// INN_S2S_44976897523887
// MI455X (gfx1250) — matrix operand order repaired and checked by reading, not compiled, not run on hardware
//
#include <hip/hip_runtime.h>
#include <hip/hip_bf16.h>
#include <string.h>

// ---------------------------------------------------------------------------
// Types for CDNA5 WMMA (wave32, v_wmma_f32_16x16x32_bf16)
// ---------------------------------------------------------------------------
typedef __bf16 bf16_t;
typedef __attribute__((ext_vector_type(16))) __bf16 v16bf;
typedef __attribute__((ext_vector_type(8)))  __bf16 v8bf;
typedef __attribute__((ext_vector_type(8)))  float  v8f;

#define DEV __device__ __forceinline__

// A-matrix 16x32 bf16 fragment (ISA 7.12.2): lanes 0-15 m=0..15 hold
// K = {k..k+7, k+16..k+23}; lanes 16-31 hold K = {k+8..k+15, k+24..k+31}.
DEV v16bf ld_afrag(const bf16_t* row, int k, int hi) {
  v16bf a;
  ((v8bf*)&a)[0] = *(const v8bf*)(row + k + 8 * hi);
  ((v8bf*)&a)[1] = *(const v8bf*)(row + k + 16 + 8 * hi);
  return a;
}
// B-matrix 32x16 bf16 fragment: lane n=lane%16; lanes 0-15 hold K={k..k+7,
// k+16..k+23}, lanes 16-31 hold K={k+8..k+15, k+24..k+31}. W stored as [N,K] row-major.
DEV v16bf ld_bfrag(const bf16_t* row, int k, int hi) {
  v16bf b;
  ((v8bf*)&b)[0] = *(const v8bf*)(row + k + 8 * hi);
  ((v8bf*)&b)[1] = *(const v8bf*)(row + k + 16 + 8 * hi);
  return b;
}
DEV v8f zacc() {
  v8f a;
#pragma unroll
  for (int i = 0; i < 8; ++i) a[i] = 0.0f;
  return a;
}
DEV float sigm(float x) { return 1.0f / (1.0f + __expf(-x)); }

// ---------------------------------------------------------------------------
// Generic batched GEMM: C[z] = act( A1[z]*W1[z]^T (+ A2[z]*W2[z]^T) + bias
//                                   (+ resid) )      A:[M,K] W:[N,K] row-major
// z decomposed as b = z / zdiv, h = z % zdiv; per-operand (b,h) strides.
// Block = 8 waves; each wave owns a 16x16 tile; block tile = 16M x 128N.
// Out-of-range M rows read from zpad (zeroed dummy row) -> branchless K loop.
// ---------------------------------------------------------------------------
struct GemmP {
  const bf16_t* A1; long lda1, sA1b, sA1h;
  const bf16_t* A2; long lda2, sA2b, sA2h;
  const bf16_t* W1; long ldw1, sW1b, sW1h;
  const bf16_t* W2; long ldw2, sW2b, sW2h;
  const float*  bias;  long sBb, sBh;
  const float*  resid; long ldr, sRb, sRh;
  float*  Cf; long ldcf, sCfb, sCfh;
  bf16_t* Cb; long ldcb, sCbb, sCbh;
  const bf16_t* zpad;   // zeroed row, >= max K elements
  int M, N, K, K2, zdiv, relu;
};

DEV v8f kloop(v8f acc, const bf16_t* arow, const bf16_t* wrow, int K, int hi) {
#pragma unroll 2
  for (int k = 0; k < K; k += 32) {
    v16bf af = ld_afrag(arow, k, hi);
    v16bf wf = ld_bfrag(wrow, k, hi);
    acc = __builtin_amdgcn_wmma_f32_16x16x32_bf16(false, af, false, wf,
                                                  (short)0, acc, false, false);
  }
  return acc;
}

__global__ __launch_bounds__(256) void wgemm(GemmP p) {
  const int wave = threadIdx.x >> 5;
  const int lane = threadIdx.x & 31;
  const int n0 = (blockIdx.x * 8 + wave) * 16;
  if (n0 >= p.N) return;
  const int m0 = blockIdx.y * 16;
  const int z  = blockIdx.z;
  const int zb = z / p.zdiv, zh = z % p.zdiv;
  const int r = lane & 15, hi = lane >> 4;
  const bool av = (m0 + r) < p.M;

  v8f acc = zacc();
  {
    const bf16_t* A = p.A1 + (long)zb * p.sA1b + (long)zh * p.sA1h;
    const bf16_t* W = p.W1 + (long)zb * p.sW1b + (long)zh * p.sW1h;
    const bf16_t* arow = av ? (A + (long)(m0 + r) * p.lda1) : p.zpad;
    const bf16_t* wrow = W + (long)(n0 + r) * p.ldw1;
    acc = kloop(acc, arow, wrow, p.K, hi);
  }
  if (p.A2) {
    const bf16_t* A = p.A2 + (long)zb * p.sA2b + (long)zh * p.sA2h;
    const bf16_t* W = p.W2 + (long)zb * p.sW2b + (long)zh * p.sW2h;
    const bf16_t* arow = av ? (A + (long)(m0 + r) * p.lda2) : p.zpad;
    const bf16_t* wrow = W + (long)(n0 + r) * p.ldw2;
    acc = kloop(acc, arow, wrow, p.K2, hi);
  }

  const float bv = p.bias ? p.bias[(long)zb * p.sBb + (long)zh * p.sBh + n0 + r] : 0.0f;
  const float* Rz = p.resid ? p.resid + (long)zb * p.sRb + (long)zh * p.sRh : nullptr;
  float*  Cfz = p.Cf ? p.Cf + (long)zb * p.sCfb + (long)zh * p.sCfh : nullptr;
  bf16_t* Cbz = p.Cb ? p.Cb + (long)zb * p.sCbb + (long)zh * p.sCbh : nullptr;
#pragma unroll
  for (int rr = 0; rr < 8; ++rr) {
    const int cr = m0 + rr + 8 * hi;          // C/D frag: row = m0+rr+8*hi
    if (cr < p.M) {
      float v = acc[rr] + bv;
      if (Rz) v += Rz[(long)cr * p.ldr + n0 + r];
      if (p.relu) v = fmaxf(v, 0.0f);
      if (Cfz) Cfz[(long)cr * p.ldcf + n0 + r] = v;
      if (Cbz) Cbz[(long)cr * p.ldcb + n0 + r] = (bf16_t)v;
    }
  }
}

// ---------------------------------------------------------------------------
// 3-tap conv1d (pad 1) as WMMA GEMM over shifted rows. X:[B*S,D] bf16,
// Wp:[3,D,D] bf16 (pre-transposed [tap][out][in]).  Out bf16 [B*S,D].
// Boundary rows read from zpad (branchless).
// ---------------------------------------------------------------------------
struct ConvP {
  const bf16_t* X; const bf16_t* Wp; const float* bias; bf16_t* Cb;
  const bf16_t* zpad;
  int S, D, relu;
};

__global__ __launch_bounds__(256) void conv3(ConvP p) {
  const int wave = threadIdx.x >> 5, lane = threadIdx.x & 31;
  const int n0 = (blockIdx.x * 8 + wave) * 16;
  if (n0 >= p.D) return;
  const int m0 = blockIdx.y * 16;
  const int r = lane & 15, hi = lane >> 4;
  const int m = m0 + r;
  const int b = m / p.S, si = m % p.S;
  v8f acc = zacc();
  for (int t = 0; t < 3; ++t) {
    const int sp = si + t - 1;
    const bool av = (sp >= 0) && (sp < p.S);
    const bf16_t* arow = av ? (p.X + ((long)b * p.S + sp) * p.D) : p.zpad;
    const bf16_t* wrow = p.Wp + (long)t * p.D * p.D + (long)(n0 + r) * p.D;
    acc = kloop(acc, arow, wrow, p.D, hi);
  }
  const float bv = p.bias[n0 + r];
#pragma unroll
  for (int rr = 0; rr < 8; ++rr) {
    const int cr = m0 + rr + 8 * hi;
    float v = acc[rr] + bv;
    if (p.relu) v = fmaxf(v, 0.0f);
    p.Cb[(long)cr * p.D + n0 + r] = (bf16_t)v;
  }
}

// ---------------------------------------------------------------------------
// Small utility kernels
// ---------------------------------------------------------------------------
__global__ void cvt_bf(const float* s, bf16_t* d, long n) {
  long i = blockIdx.x * 256L + threadIdx.x;
  if (i < n) d[i] = (bf16_t)s[i];
}
__global__ void zero_f32(float* p, long n) {
  long i = blockIdx.x * 256L + threadIdx.x;
  if (i < n) p[i] = 0.0f;
}
__global__ void zero_bf(bf16_t* p, long n) {
  long i = blockIdx.x * 256L + threadIdx.x;
  if (i < n) p[i] = (bf16_t)0.0f;
}
__global__ void add_vec(const float* a, const float* b, float* c, long n) {
  long i = blockIdx.x * 256L + threadIdx.x;
  if (i < n) c[i] = a[i] + b[i];
}
// conv_w [D,D,3] (o,i,t) -> wp [3,D,D] bf16
__global__ void prep_convw(const float* w, bf16_t* wp, int D) {
  long i = blockIdx.x * 256L + threadIdx.x;
  long total = (long)D * D * 3;
  if (i >= total) return;
  int t = (int)(i % 3);
  long oi = i / 3;
  long o = oi / D, ii = oi % D;
  wp[((long)t * D + o) * D + ii] = (bf16_t)w[i];
}
// embedding gather: one block per row
__global__ void gather_emb(const int* tok, const float* emb, bf16_t* ob,
                           float* of, int D) {
  const long row = blockIdx.x;
  const float* src = emb + (long)tok[row] * D;
  for (int d = threadIdx.x; d < D; d += blockDim.x) {
    float v = src[d];
    if (ob) ob[row * (long)D + d] = (bf16_t)v;
    if (of) of[row * (long)D + d] = v;
  }
}
// V [B,S,D] bf16 -> Vt [B,NH,hd,S] bf16
__global__ void vtrans(const bf16_t* V, bf16_t* Vt, int B_, int S_, int D_,
                       int NH_, int hd_) {
  long i = blockIdx.x * 256L + threadIdx.x;
  long total = (long)B_ * S_ * D_;
  if (i >= total) return;
  int d = (int)(i % D_);
  long bs = i / D_;
  int s = (int)(bs % S_);
  int b = (int)(bs / S_);
  int h = d / hd_, dd = d % hd_;
  Vt[(((long)b * NH_ + h) * hd_ + dd) * S_ + s] = V[i];
}
// row softmax (scale folded in), fp32 in -> bf16 probs
__global__ void softmax_rows(const float* X, bf16_t* P, int cols, float scale) {
  const long row = blockIdx.x;
  const float* x = X + row * (long)cols;
  bf16_t* o = P + row * (long)cols;
  __shared__ float red[256];
  float mx = -1e30f;
  for (int c = threadIdx.x; c < cols; c += 256) mx = fmaxf(mx, x[c] * scale);
  red[threadIdx.x] = mx; __syncthreads();
  for (int st = 128; st > 0; st >>= 1) {
    if (threadIdx.x < st) red[threadIdx.x] = fmaxf(red[threadIdx.x], red[threadIdx.x + st]);
    __syncthreads();
  }
  mx = red[0]; __syncthreads();
  float sum = 0.0f;
  for (int c = threadIdx.x; c < cols; c += 256) sum += __expf(x[c] * scale - mx);
  red[threadIdx.x] = sum; __syncthreads();
  for (int st = 128; st > 0; st >>= 1) {
    if (threadIdx.x < st) red[threadIdx.x] += red[threadIdx.x + st];
    __syncthreads();
  }
  const float inv = 1.0f / red[0];
  for (int c = threadIdx.x; c < cols; c += 256)
    o[c] = (bf16_t)(__expf(x[c] * scale - mx) * inv);
}
// LSTM cell pointwise: gates [N,B,4D] -> c, h_bf [N,B,D], stack [B,N,D]
__global__ void lstm_cell(const float* gates, float* c, bf16_t* h_bf,
                          float* stack_f, bf16_t* stack_bf, int NB, int Bb, int D_) {
  long i = blockIdx.x * 256L + threadIdx.x;
  long total = (long)NB * Bb * D_;
  if (i >= total) return;
  long nb = i / D_;
  int d = (int)(i % D_);
  int n = (int)(nb / Bb), b = (int)(nb % Bb);
  const float* g = gates + nb * (long)(4 * D_);
  float ig = sigm(g[d]);
  float fg = sigm(g[D_ + d]);
  float gg = tanhf(g[2 * D_ + d]);
  float og = sigm(g[3 * D_ + d]);
  float cn = fg * c[i] + ig * gg;
  c[i] = cn;
  float h = og * tanhf(cn);
  h_bf[i] = (bf16_t)h;
  long si = ((long)b * NB + n) * D_ + d;
  stack_f[si] = h;
  stack_bf[si] = (bf16_t)h;
}
// mix attention (Lq=1, 8 keys): q [B,D] f32, kv [B*8,2048] f32 (k|v) -> ctx bf16 [B,D]
__global__ void mixattn(const float* q, const float* kv, bf16_t* out, int D_) {
  const int b = blockIdx.x;
  __shared__ float sc[32];
  __shared__ float pr[32];
  const int tid = threadIdx.x;
  if (tid < 32) {
    int h = tid >> 3, j = tid & 7;
    const float* qp = q + (long)b * D_ + h * 256;
    const float* kp = kv + ((long)(b * 8 + j)) * 2048 + h * 256;
    float s = 0.0f;
    for (int d = 0; d < 256; ++d) s += qp[d] * kp[d];
    sc[tid] = s * 0.0625f;  // 1/sqrt(256)
  }
  __syncthreads();
  if (tid < 4) {
    float m = -1e30f;
    for (int j = 0; j < 8; ++j) m = fmaxf(m, sc[tid * 8 + j]);
    float sm = 0.0f;
    for (int j = 0; j < 8; ++j) { float e = __expf(sc[tid * 8 + j] - m); pr[tid * 8 + j] = e; sm += e; }
    float inv = 1.0f / sm;
    for (int j = 0; j < 8; ++j) pr[tid * 8 + j] *= inv;
  }
  __syncthreads();
  for (int d = tid; d < D_; d += 256) {
    int h = d >> 8;
    float a = 0.0f;
    for (int j = 0; j < 8; ++j)
      a += pr[h * 8 + j] * kv[((long)(b * 8 + j)) * 2048 + 1024 + d];
    out[(long)b * D_ + d] = (bf16_t)a;
  }
}
// LayerNorm rows of D, fp32 in -> bf16 out
__global__ void lnorm(const float* x, const float* g, const float* be,
                      bf16_t* o, int D_) {
  const long row = blockIdx.x;
  const float* xr = x + row * (long)D_;
  __shared__ float red[256];
  float s = 0.0f;
  for (int d = threadIdx.x; d < D_; d += 256) s += xr[d];
  red[threadIdx.x] = s; __syncthreads();
  for (int st = 128; st > 0; st >>= 1) {
    if (threadIdx.x < st) red[threadIdx.x] += red[threadIdx.x + st];
    __syncthreads();
  }
  const float mu = red[0] / D_; __syncthreads();
  float v = 0.0f;
  for (int d = threadIdx.x; d < D_; d += 256) { float t = xr[d] - mu; v += t * t; }
  red[threadIdx.x] = v; __syncthreads();
  for (int st = 128; st > 0; st >>= 1) {
    if (threadIdx.x < st) red[threadIdx.x] += red[threadIdx.x + st];
    __syncthreads();
  }
  const float inv = rsqrtf(red[0] / D_ + 1e-5f);
  for (int d = threadIdx.x; d < D_; d += 256)
    o[row * (long)D_ + d] = (bf16_t)((xr[d] - mu) * inv * g[d] + be[d]);
}

// ---------------------------------------------------------------------------
// Host side
// ---------------------------------------------------------------------------
namespace {
constexpr int B = 8, S = 512, T = 256, D = 1024, NL = 8;
constexpr int Va = 2048, NH = 4, HD = 256;

inline GemmP gp0(const bf16_t* zpad) {
  GemmP p; memset(&p, 0, sizeof(p)); p.zdiv = 1; p.zpad = zpad; return p;
}
inline void launch_gemm(hipStream_t st, const GemmP& p, int Z) {
  dim3 g((unsigned)((p.N + 127) / 128), (unsigned)((p.M + 15) / 16), (unsigned)Z);
  wgemm<<<g, 256, 0, st>>>(p);
}
inline unsigned gr(long n) { return (unsigned)((n + 255) / 256); }
}  // namespace

extern "C" void kernel_launch(void* const* d_in, const int* in_sizes, int n_in,
                              void* d_out, int out_size, void* d_ws, size_t ws_size,
                              hipStream_t stream) {
  (void)in_sizes; (void)n_in; (void)out_size; (void)ws_size;
  const int*   sem_tok   = (const int*)d_in[0];
  const int*   aud_tok   = (const int*)d_in[1];
  const float* sem_emb   = (const float*)d_in[2];
  const float* conv1_w   = (const float*)d_in[3];
  const float* conv1_b   = (const float*)d_in[4];
  const float* conv2_w   = (const float*)d_in[5];
  const float* conv2_b   = (const float*)d_in[6];
  const float* proj_w    = (const float*)d_in[7];
  const float* proj_b    = (const float*)d_in[8];
  const float* audio_emb = (const float*)d_in[9];
  const float* lstm_Wih  = (const float*)d_in[10];
  const float* lstm_Whh  = (const float*)d_in[11];
  const float* lstm_bih  = (const float*)d_in[12];
  const float* lstm_bhh  = (const float*)d_in[13];
  const float* cross_in_w  = (const float*)d_in[14];
  const float* cross_in_b  = (const float*)d_in[15];
  const float* cross_out_w = (const float*)d_in[16];
  const float* cross_out_b = (const float*)d_in[17];
  const float* mix_in_w  = (const float*)d_in[18];
  const float* mix_in_b  = (const float*)d_in[19];
  const float* mix_out_w = (const float*)d_in[20];
  const float* mix_out_b = (const float*)d_in[21];
  const float* ln_g = (const float*)d_in[22];
  const float* ln_b = (const float*)d_in[23];
  const float* head_w = (const float*)d_in[24];
  const float* head_b = (const float*)d_in[25];
  float* out = (float*)d_out;

  // ---- workspace bump allocator ----
  char* wsb = (char*)d_ws;
  size_t off = 0;
  auto alloc = [&](size_t bytes) -> void* {
    off = (off + 255) & ~(size_t)255;
    void* p = wsb + off;
    off += bytes;
    return p;
  };
  const long DD = (long)D * D;
  bf16_t* zpad  = (bf16_t*)alloc((long)D * 2);     // zeroed dummy row (max K)
  bf16_t* c1wp  = (bf16_t*)alloc(3 * DD * 2);
  bf16_t* c2wp  = (bf16_t*)alloc(3 * DD * 2);
  bf16_t* projw = (bf16_t*)alloc(DD * 2);
  bf16_t* ciw   = (bf16_t*)alloc(3 * DD * 2);
  bf16_t* cow   = (bf16_t*)alloc(DD * 2);
  bf16_t* miw   = (bf16_t*)alloc(3 * DD * 2);
  bf16_t* mow   = (bf16_t*)alloc(DD * 2);
  bf16_t* headw = (bf16_t*)alloc((long)Va * D * 2);
  bf16_t* wih   = (bf16_t*)alloc((long)NL * 4 * DD * 2);
  bf16_t* whh   = (bf16_t*)alloc((long)NL * 4 * DD * 2);
  float*  bcomb = (float*)alloc((long)NL * 4 * D * 4);

  bf16_t* semx  = (bf16_t*)alloc((long)B * S * D * 2);
  bf16_t* c1o   = (bf16_t*)alloc((long)B * S * D * 2);
  bf16_t* c2o   = (bf16_t*)alloc((long)B * S * D * 2);
  bf16_t* memb  = (bf16_t*)alloc((long)B * S * D * 2);
  bf16_t* kc    = (bf16_t*)alloc((long)B * S * D * 2);
  bf16_t* vc    = (bf16_t*)alloc((long)B * S * D * 2);
  bf16_t* vt    = (bf16_t*)alloc((long)B * S * D * 2);
  float*  xemb_f = (float*)alloc((long)B * T * D * 4);
  bf16_t* xemb_b = (bf16_t*)alloc((long)B * T * D * 2);
  float*  gates  = (float*)alloc((long)NL * B * 4 * D * 4);
  float*  c_st   = (float*)alloc((long)NL * B * D * 4);
  bf16_t* h_bf   = (bf16_t*)alloc((long)NL * B * D * 2);
  float*  stk_f  = (float*)alloc((long)B * NL * D * 4);
  bf16_t* stk_b  = (bf16_t*)alloc((long)B * NL * D * 2);
  bf16_t* qc     = (bf16_t*)alloc((long)B * 8 * D * 2);
  float*  scf    = (float*)alloc((long)B * NH * 8 * S * 4);
  bf16_t* pbf    = (bf16_t*)alloc((long)B * NH * 8 * S * 2);
  bf16_t* ctxb   = (bf16_t*)alloc((long)B * 8 * D * 2);
  bf16_t* enrb   = (bf16_t*)alloc((long)B * 8 * D * 2);
  float*  mq     = (float*)alloc((long)B * D * 4);
  float*  mkv    = (float*)alloc((long)B * 8 * 2 * D * 4);
  bf16_t* ctx2   = (bf16_t*)alloc((long)B * D * 2);
  float*  preln  = (float*)alloc((long)B * D * 4);
  bf16_t* mixed  = (bf16_t*)alloc((long)B * D * 2);

  // ---- one-time prep: bf16 weights, conv weight transpose, combined bias ----
  zero_bf<<<gr(D), 256, 0, stream>>>(zpad, D);
  prep_convw<<<gr(3 * DD), 256, 0, stream>>>(conv1_w, c1wp, D);
  prep_convw<<<gr(3 * DD), 256, 0, stream>>>(conv2_w, c2wp, D);
  cvt_bf<<<gr(DD), 256, 0, stream>>>(proj_w, projw, DD);
  cvt_bf<<<gr(3 * DD), 256, 0, stream>>>(cross_in_w, ciw, 3 * DD);
  cvt_bf<<<gr(DD), 256, 0, stream>>>(cross_out_w, cow, DD);
  cvt_bf<<<gr(3 * DD), 256, 0, stream>>>(mix_in_w, miw, 3 * DD);
  cvt_bf<<<gr(DD), 256, 0, stream>>>(mix_out_w, mow, DD);
  cvt_bf<<<gr((long)Va * D), 256, 0, stream>>>(head_w, headw, (long)Va * D);
  cvt_bf<<<gr((long)NL * 4 * DD), 256, 0, stream>>>(lstm_Wih, wih, (long)NL * 4 * DD);
  cvt_bf<<<gr((long)NL * 4 * DD), 256, 0, stream>>>(lstm_Whh, whh, (long)NL * 4 * DD);
  add_vec<<<gr((long)NL * 4 * D), 256, 0, stream>>>(lstm_bih, lstm_bhh, bcomb, (long)NL * 4 * D);

  // ---- initial state (must re-init every call: graph replays) ----
  zero_f32<<<gr((long)NL * B * D), 256, 0, stream>>>(c_st, (long)NL * B * D);
  zero_bf<<<gr((long)NL * B * D), 256, 0, stream>>>(h_bf, (long)NL * B * D);

  // ---- embeddings ----
  gather_emb<<<B * S, 256, 0, stream>>>(sem_tok, sem_emb, semx, nullptr, D);
  gather_emb<<<B * T, 256, 0, stream>>>(aud_tok, audio_emb, xemb_b, xemb_f, D);

  // ---- encoder: conv1(relu) -> conv2 -> proj ----
  {
    ConvP cp; cp.X = semx; cp.Wp = c1wp; cp.bias = conv1_b; cp.Cb = c1o;
    cp.zpad = zpad; cp.S = S; cp.D = D; cp.relu = 1;
    conv3<<<dim3(D / 128, (B * S) / 16, 1), 256, 0, stream>>>(cp);
    cp.X = c1o; cp.Wp = c2wp; cp.bias = conv2_b; cp.Cb = c2o; cp.relu = 0;
    conv3<<<dim3(D / 128, (B * S) / 16, 1), 256, 0, stream>>>(cp);
  }
  {
    GemmP p = gp0(zpad);
    p.A1 = c2o; p.lda1 = D; p.W1 = projw; p.ldw1 = D; p.bias = proj_b;
    p.Cb = memb; p.ldcb = D; p.M = B * S; p.N = D; p.K = D;
    launch_gemm(stream, p, 1);
  }
  // ---- hoisted K / V projections of memory (loop-invariant!) ----
  {
    GemmP p = gp0(zpad);
    p.A1 = memb; p.lda1 = D; p.W1 = ciw + DD; p.ldw1 = D; p.bias = cross_in_b + D;
    p.Cb = kc; p.ldcb = D; p.M = B * S; p.N = D; p.K = D;
    launch_gemm(stream, p, 1);
    p.W1 = ciw + 2 * DD; p.bias = cross_in_b + 2 * D; p.Cb = vc;
    launch_gemm(stream, p, 1);
  }
  vtrans<<<gr((long)B * S * D), 256, 0, stream>>>(vc, vt, B, S, D, NH, HD);

  // ---- sequential scan over T steps ----
  for (int t = 0; t < T; ++t) {
    // LSTM gates: xt@Wih^T + h@Whh^T + (bih+bhh), batched over NL layers
    {
      GemmP p = gp0(zpad);
      p.A1 = xemb_b + (long)t * D; p.lda1 = (long)T * D;
      p.A2 = h_bf; p.lda2 = D; p.sA2b = (long)B * D;
      p.W1 = wih; p.ldw1 = D; p.sW1b = 4 * DD;
      p.W2 = whh; p.ldw2 = D; p.sW2b = 4 * DD;
      p.bias = bcomb; p.sBb = 4L * D;
      p.Cf = gates; p.ldcf = 4L * D; p.sCfb = (long)B * 4 * D;
      p.M = B; p.N = 4 * D; p.K = D; p.K2 = D;
      launch_gemm(stream, p, NL);
    }
    lstm_cell<<<gr((long)NL * B * D), 256, 0, stream>>>(gates, c_st, h_bf, stk_f,
                                                        stk_b, NL, B, D);
    // cross-attn q projection of stack [B*8, D]
    {
      GemmP p = gp0(zpad);
      p.A1 = stk_b; p.lda1 = D; p.W1 = ciw; p.ldw1 = D; p.bias = cross_in_b;
      p.Cb = qc; p.ldcb = D; p.M = B * 8; p.N = D; p.K = D;
      launch_gemm(stream, p, 1);
    }
    // scores[b,h] = qh[8,hd] @ kh[S,hd]^T   (Z = B*NH)
    {
      GemmP p = gp0(zpad);
      p.zdiv = NH;
      p.A1 = qc; p.lda1 = D; p.sA1b = 8L * D; p.sA1h = HD;
      p.W1 = kc; p.ldw1 = D; p.sW1b = (long)S * D; p.sW1h = HD;
      p.Cf = scf; p.ldcf = S; p.sCfb = (long)NH * 8 * S; p.sCfh = 8L * S;
      p.M = 8; p.N = S; p.K = HD;
      launch_gemm(stream, p, B * NH);
    }
    softmax_rows<<<B * NH * 8, 256, 0, stream>>>(scf, pbf, S, 0.0625f);
    // ctx[b,h] = P[8,S] @ Vt[b,h][hd,S]^T
    {
      GemmP p = gp0(zpad);
      p.zdiv = NH;
      p.A1 = pbf; p.lda1 = S; p.sA1b = (long)NH * 8 * S; p.sA1h = 8L * S;
      p.W1 = vt; p.ldw1 = S; p.sW1b = (long)NH * HD * S; p.sW1h = (long)HD * S;
      p.Cb = ctxb; p.ldcb = D; p.sCbb = 8L * D; p.sCbh = HD;
      p.M = 8; p.N = HD; p.K = S;
      launch_gemm(stream, p, B * NH);
    }
    // cross out-proj + residual stack -> enriched
    {
      GemmP p = gp0(zpad);
      p.A1 = ctxb; p.lda1 = D; p.W1 = cow; p.ldw1 = D; p.bias = cross_out_b;
      p.resid = stk_f; p.ldr = D;
      p.Cb = enrb; p.ldcb = D; p.M = B * 8; p.N = D; p.K = D;
      launch_gemm(stream, p, 1);
    }
    // mix q projection of xt
    {
      GemmP p = gp0(zpad);
      p.A1 = xemb_b + (long)t * D; p.lda1 = (long)T * D;
      p.W1 = miw; p.ldw1 = D; p.bias = mix_in_b;
      p.Cf = mq; p.ldcf = D; p.M = B; p.N = D; p.K = D;
      launch_gemm(stream, p, 1);
    }
    // mix k|v projections of enriched (one GEMM, N=2D covers wk and wv rows)
    {
      GemmP p = gp0(zpad);
      p.A1 = enrb; p.lda1 = D; p.W1 = miw + DD; p.ldw1 = D; p.bias = mix_in_b + D;
      p.Cf = mkv; p.ldcf = 2L * D; p.M = B * 8; p.N = 2 * D; p.K = D;
      launch_gemm(stream, p, 1);
    }
    mixattn<<<B, 256, 0, stream>>>(mq, mkv, ctx2, D);
    // mix out-proj + residual xt
    {
      GemmP p = gp0(zpad);
      p.A1 = ctx2; p.lda1 = D; p.W1 = mow; p.ldw1 = D; p.bias = mix_out_b;
      p.resid = xemb_f + (long)t * D; p.ldr = (long)T * D;
      p.Cf = preln; p.ldcf = D; p.M = B; p.N = D; p.K = D;
      launch_gemm(stream, p, 1);
    }
    lnorm<<<B, 256, 0, stream>>>(preln, ln_g, ln_b, mixed, D);
    // head: out[b,t,:] = mixed @ head_w^T + head_b
    {
      GemmP p = gp0(zpad);
      p.A1 = mixed; p.lda1 = D; p.W1 = headw; p.ldw1 = D; p.bias = head_b;
      p.Cf = out + (long)t * Va; p.ldcf = (long)T * Va;
      p.M = B; p.N = Va; p.K = D;
      launch_gemm(stream, p, 1);
    }
  }
}
